// MultiHeadQDGAT_12043088298063
// MI455X (gfx1250) — matrix operand order repaired and checked by reading, not compiled, not run on hardware
//
#include <hip/hip_runtime.h>
#include <hip/hip_bf16.h>
#include <math.h>

#define B_ 4
#define N_ 1024
#define D_ 512
#define NH_ 8
#define H_ 64
#define T_ 2
#define NTYPES_ 8
#define D2_ 1024
#define M_ (B_*N_)   // 4096

typedef __bf16 bf16_t;
typedef bf16_t v16bf __attribute__((ext_vector_type(16)));
typedef bf16_t v8bf  __attribute__((ext_vector_type(8)));
typedef float  v8f   __attribute__((ext_vector_type(8)));

static __device__ __forceinline__ bf16_t f2bf(float f) { return (bf16_t)f; }

// ---------------------------------------------------------------- utilities
__global__ void k_copy_f32(const float* __restrict__ src, float* __restrict__ dst, int n) {
    int i = blockIdx.x * blockDim.x + threadIdx.x;
    if (i < n) dst[i] = src[i];
}

// Pack weight W[K,Nout] (f32 row-major) into WMMA B-fragment layout, bf16.
// Fragment for (n-tile nt, k-tile kt): 32 lanes x 16 contiguous bf16.
// lane: column = nt*16 + lane%16 ; element e: k = kt*32 + (lane<16?0:16) + e
__global__ void k_pack_b(const float* __restrict__ W, bf16_t* __restrict__ Bp, int K, int Nout) {
    int tid = blockIdx.x * blockDim.x + threadIdx.x;
    int KT = K >> 5;
    int total = KT * (Nout >> 4) * 32;
    if (tid >= total) return;
    int lane = tid & 31;
    int t2 = tid >> 5;
    int kt = t2 % KT;
    int nt = t2 / KT;
    int n = (nt << 4) + (lane & 15);
    int kbase = (kt << 5) + ((lane < 16) ? 0 : 16);
    bf16_t* out = Bp + ((size_t)((nt * KT + kt) * 32 + lane) << 4);
#pragma unroll
    for (int e = 0; e < 16; ++e)
        out[e] = f2bf(W[(size_t)(kbase + e) * Nout + n]);
}

// h1[b] = elu(question_embedding[b] @ W_fc)   [B, 2D]
__global__ void k_h1(const float* __restrict__ q, const float* __restrict__ Wfc,
                     float* __restrict__ h1) {
    __shared__ float qs[D_];
    int b = blockIdx.x;
    for (int i = threadIdx.x; i < D_; i += blockDim.x) qs[i] = q[b * D_ + i];
    __syncthreads();
    for (int j = threadIdx.x; j < D2_; j += blockDim.x) {
        float acc = 0.f;
        for (int k = 0; k < D_; ++k) acc += qs[k] * Wfc[(size_t)k * D2_ + j];
        h1[b * D2_ + j] = acc > 0.f ? acc : (__expf(acc) - 1.f);
    }
}

// per (t,b): m_t = h1 @ w_dc[t]; gates g{q,k,v}[t,b] = m_t @ W_{q,k,v}c
__global__ void k_gates(const float* __restrict__ h1, const float* __restrict__ wdc,
                        const float* __restrict__ Wqc, const float* __restrict__ Wkc,
                        const float* __restrict__ Wvc,
                        float* __restrict__ gq, float* __restrict__ gk, float* __restrict__ gv) {
    __shared__ float hs[D2_];
    __shared__ float ms[D_];
    int t = blockIdx.x / B_;
    int b = blockIdx.x % B_;
    for (int i = threadIdx.x; i < D2_; i += blockDim.x) hs[i] = h1[b * D2_ + i];
    __syncthreads();
    const float* wd = wdc + (size_t)t * D2_ * D_;
    for (int j = threadIdx.x; j < D_; j += blockDim.x) {
        float acc = 0.f;
        for (int k = 0; k < D2_; ++k) acc += hs[k] * wd[(size_t)k * D_ + j];
        ms[j] = acc;
    }
    __syncthreads();
    size_t off = ((size_t)t * B_ + b) * D_;
    for (int j = threadIdx.x; j < D_; j += blockDim.x) {
        float aq = 0.f, ak = 0.f, av = 0.f;
        for (int k = 0; k < D_; ++k) {
            float m = ms[k];
            aq += m * Wqc[(size_t)k * D_ + j];
            ak += m * Wkc[(size_t)k * D_ + j];
            av += m * Wvc[(size_t)k * D_ + j];
        }
        gq[off + j] = aq; gk[off + j] = ak; gv[off + j] = av;
    }
}

// Compress 9 disjoint int32 masks [*,B,N,N] into int8 code: -1 none, 0 entity, 1..8 type
__global__ void k_codes(const int* __restrict__ me, const int* __restrict__ mt,
                        signed char* __restrict__ codes) {
    size_t i = (size_t)blockIdx.x * blockDim.x + threadIdx.x;
    const size_t total = (size_t)B_ * N_ * N_;
    if (i >= total) return;
    int c = -1;
    if (me[i] != 0) c = 0;
#pragma unroll
    for (int k = 0; k < NTYPES_; ++k)
        if (mt[(size_t)k * total + i] != 0) c = k + 1;
    codes[i] = (signed char)c;
}

// concat(lo[.,512], hi[.,512]) -> bf16 [4096, 1024]
__global__ void k_cat_bf16(const float* __restrict__ lo, const float* __restrict__ hi,
                           bf16_t* __restrict__ out) {
    size_t i = (size_t)blockIdx.x * blockDim.x + threadIdx.x;
    const size_t total = (size_t)M_ * D2_;
    if (i >= total) return;
    int col = (int)(i & (D2_ - 1));
    size_t row = i >> 10;
    float v = (col < D_) ? lo[row * D_ + col] : hi[row * D_ + (col - D_)];
    out[i] = f2bf(v);
}

// ---------------------------------------------------------------- bf16 WMMA GEMM
// C[M,Nout] = A[M,K](bf16,row-major) x Bp(packed). Wave computes 16x64 tile.
// Optional per-(batch,col) gate multiplier (rowsPerB rows per batch).
__global__ void k_gemm_bf16(const bf16_t* __restrict__ A, const bf16_t* __restrict__ Bp,
                            float* __restrict__ C, int Mtiles, int KT, int Nout,
                            const float* __restrict__ gate, int rowsPerB) {
    int lane = threadIdx.x & 31;
    int tile = blockIdx.x * (blockDim.x >> 5) + (threadIdx.x >> 5);
    int NT4 = Nout >> 6;
    if (tile >= Mtiles * NT4) return;
    int mt = tile % Mtiles;
    int nt0 = (tile / Mtiles) << 2;
    int K = KT << 5;
    int hi = (lane < 16) ? 0 : 8;     // C row-offset for upper half-wave (A K-offset is 2*hi)
    int row = (mt << 4) + (lane & 15);
    const bf16_t* arow = A + (size_t)row * K;
    const size_t bstride = (size_t)KT * 32 * 16;  // elements between consecutive n-tiles
    v8f acc0 = {}, acc1 = {}, acc2 = {}, acc3 = {};
    for (int kt = 0; kt < KT; ++kt) {
        int k0 = kt << 5;
        v8bf a_lo = *(const v8bf*)(arow + k0 + 2 * hi);
        v8bf a_hi = *(const v8bf*)(arow + k0 + 2 * hi + 8);
        v16bf a;
#pragma unroll
        for (int e = 0; e < 8; ++e) { a[e] = a_lo[e]; a[e + 8] = a_hi[e]; }
        const bf16_t* bbase = Bp + (((size_t)nt0 * KT + kt) * 32 + lane) * 16;
        v16bf b0 = *(const v16bf*)(bbase);
        v16bf b1 = *(const v16bf*)(bbase + bstride);
        v16bf b2 = *(const v16bf*)(bbase + 2 * bstride);
        v16bf b3 = *(const v16bf*)(bbase + 3 * bstride);
        acc0 = __builtin_amdgcn_wmma_f32_16x16x32_bf16(false, a, false, b0, (short)0, acc0, false, false);
        acc1 = __builtin_amdgcn_wmma_f32_16x16x32_bf16(false, a, false, b1, (short)0, acc1, false, false);
        acc2 = __builtin_amdgcn_wmma_f32_16x16x32_bf16(false, a, false, b2, (short)0, acc2, false, false);
        acc3 = __builtin_amdgcn_wmma_f32_16x16x32_bf16(false, a, false, b3, (short)0, acc3, false, false);
    }
    int coln = lane & 15;
    v8f accs[4] = {acc0, acc1, acc2, acc3};
#pragma unroll
    for (int i = 0; i < 4; ++i) {
        int col = ((nt0 + i) << 4) + coln;
#pragma unroll
        for (int r = 0; r < 8; ++r) {
            int m = (mt << 4) + r + hi;
            float v = accs[i][r];
            if (gate) v *= gate[(size_t)(m / rowsPerB) * Nout + col];
            C[(size_t)m * Nout + col] = v;
        }
    }
}

// ---------------------------------------------------------------- row LayerNorm
__global__ void k_ln(const float* __restrict__ X, const float* __restrict__ g,
                     const float* __restrict__ bta, float* __restrict__ outf,
                     bf16_t* __restrict__ outb, int C) {
    __shared__ float red[256];
    int row = blockIdx.x;
    const float* x = X + (size_t)row * C;
    int nper = C >> 8;   // 2 or 4
    float xr[4];
    float s = 0.f;
    for (int i = 0; i < nper; ++i) { xr[i] = x[threadIdx.x + (i << 8)]; s += xr[i]; }
    red[threadIdx.x] = s; __syncthreads();
    for (int o = 128; o > 0; o >>= 1) {
        if (threadIdx.x < o) red[threadIdx.x] += red[threadIdx.x + o];
        __syncthreads();
    }
    float mean = red[0] / C;
    __syncthreads();
    float s2 = 0.f;
    for (int i = 0; i < nper; ++i) { float d = xr[i] - mean; s2 += d * d; }
    red[threadIdx.x] = s2; __syncthreads();
    for (int o = 128; o > 0; o >>= 1) {
        if (threadIdx.x < o) red[threadIdx.x] += red[threadIdx.x + o];
        __syncthreads();
    }
    float rstd = rsqrtf(red[0] / C + 1e-12f);
    for (int i = 0; i < nper; ++i) {
        int c = threadIdx.x + (i << 8);
        float y = (xr[i] - mean) * rstd * g[c] + bta[c];
        if (outf) outf[(size_t)row * C + c] = y;
        if (outb) outb[(size_t)row * C + c] = f2bf(y);
    }
}

// a_src/a_dst[m,b,h,n] = dot(x{q,k}[b,n,h*64:], W_att[m, sel*64:])
__global__ void k_att_proj(const float* __restrict__ xq, const float* __restrict__ xk,
                           const float* __restrict__ Watt,
                           float* __restrict__ asrc, float* __restrict__ adst) {
    int tid = blockIdx.x * blockDim.x + threadIdx.x;
    const int per = 9 * B_ * NH_ * N_;
    if (tid >= 2 * per) return;
    int sel = tid / per;
    int r = tid % per;
    int m = r / (B_ * NH_ * N_);
    int r2 = r % (B_ * NH_ * N_);
    int b = r2 / (NH_ * N_);
    int r3 = r2 % (NH_ * N_);
    int h = r3 / N_;
    int n = r3 % N_;
    const float* x = (sel == 0) ? xq : xk;
    const float* w = Watt + m * (2 * H_) + sel * H_;
    const float* xrow = x + ((size_t)b * N_ + n) * D_ + h * H_;
    float acc = 0.f;
#pragma unroll
    for (int k = 0; k < H_; ++k) acc += xrow[k] * w[k];
    ((sel == 0) ? asrc : adst)[r] = acc;
}

// transpose xv [B,N,D] f32 -> xvt [B,NH,64,N] bf16 (V in column-streamable layout)
__global__ void k_xvt(const float* __restrict__ xv, bf16_t* __restrict__ xvt) {
    size_t i = (size_t)blockIdx.x * blockDim.x + threadIdx.x;
    const size_t total = (size_t)B_ * NH_ * H_ * N_;
    if (i >= total) return;
    int j = (int)(i & (N_ - 1));
    size_t r = i >> 10;
    int d = (int)(r & (H_ - 1));
    size_t r2 = r >> 6;
    int h = (int)(r2 & (NH_ - 1));
    int b = (int)(r2 >> 3);
    xvt[i] = f2bf(xv[((size_t)b * N_ + j) * D_ + h * H_ + d]);
}

// ---------------------------------------------------------------- flash attention
// one wave per (b, h, 16-row i-tile); stream j in chunks of 32; WMMA P(16x32)xV(32x64)
__global__ void k_attn(const signed char* __restrict__ codes, const float* __restrict__ asrc,
                       const float* __restrict__ adst, const bf16_t* __restrict__ xvt,
                       float* __restrict__ X) {
    int lane = threadIdx.x & 31;
    int wave = blockIdx.x * (blockDim.x >> 5) + (threadIdx.x >> 5);
    int itile = wave & 63;
    int h = (wave >> 6) & 7;
    int b = wave >> 9;
    if (b >= B_) return;
    int lrow16 = lane & 15;
    int irow = (itile << 4) + lrow16;
    int hi8 = (lane < 16) ? 0 : 8;
    int jsel = (lane < 16) ? 0 : 16;
    float as[9];
#pragma unroll
    for (int m = 0; m < 9; ++m)
        as[m] = asrc[(((size_t)m * B_ + b) * NH_ + h) * N_ + irow];
    const signed char* crow = codes + ((size_t)b * N_ + irow) * N_;
    const bf16_t* vb_base = xvt + ((size_t)b * NH_ + h) * H_ * N_;
    float mrun = 0.f, lrun = 0.f;   // running max starts at 0 (matches masked_softmax)
    v8f acc0 = {}, acc1 = {}, acc2 = {}, acc3 = {};
    for (int j0 = 0; j0 < N_; j0 += 32) {
        unsigned long long q0 = *(const unsigned long long*)(crow + j0 + jsel);
        unsigned long long q1 = *(const unsigned long long*)(crow + j0 + 8 + jsel);
        float s[16];
        float mloc = -3.0e38f;
#pragma unroll
        for (int e = 0; e < 16; ++e) {
            int c = (int)(signed char)(((e < 8 ? q0 : q1) >> ((e & 7) * 8)) & 0xFF);
            float v = -3.0e38f;
            if (c >= 0) {
                int j = j0 + jsel + e;
                v = as[c] + adst[(((size_t)c * B_ + b) * NH_ + h) * N_ + j];
                v = v > 0.f ? v : 0.01f * v;     // leaky_relu
                mloc = fmaxf(mloc, v);
            }
            s[e] = v;
        }
        mloc = fmaxf(mloc, __shfl_xor(mloc, 16, 32));
        float mnew = fmaxf(mrun, mloc);
        float alpha = __expf(mrun - mnew);
        float psum = 0.f;
        v16bf pfrag;
#pragma unroll
        for (int e = 0; e < 16; ++e) {
            float p = (s[e] > -1.0e38f) ? __expf(s[e] - mnew) : 0.f;
            psum += p;
            pfrag[e] = f2bf(p);
        }
        psum += __shfl_xor(psum, 16, 32);
        lrun = lrun * alpha + psum;
        mrun = mnew;
        float al[8];
#pragma unroll
        for (int r = 0; r < 8; ++r) al[r] = __shfl(alpha, r + hi8, 32);
#pragma unroll
        for (int r = 0; r < 8; ++r) {
            acc0[r] *= al[r]; acc1[r] *= al[r]; acc2[r] *= al[r]; acc3[r] *= al[r];
        }
        const bf16_t* vcol = vb_base + (size_t)lrow16 * N_ + j0 + ((lane < 16) ? 0 : 16);
        v16bf b0 = *(const v16bf*)(vcol);
        v16bf b1 = *(const v16bf*)(vcol + 16 * N_);
        v16bf b2 = *(const v16bf*)(vcol + 32 * N_);
        v16bf b3 = *(const v16bf*)(vcol + 48 * N_);
        acc0 = __builtin_amdgcn_wmma_f32_16x16x32_bf16(false, pfrag, false, b0, (short)0, acc0, false, false);
        acc1 = __builtin_amdgcn_wmma_f32_16x16x32_bf16(false, pfrag, false, b1, (short)0, acc1, false, false);
        acc2 = __builtin_amdgcn_wmma_f32_16x16x32_bf16(false, pfrag, false, b2, (short)0, acc2, false, false);
        acc3 = __builtin_amdgcn_wmma_f32_16x16x32_bf16(false, pfrag, false, b3, (short)0, acc3, false, false);
    }
    float linv[8];
#pragma unroll
    for (int r = 0; r < 8; ++r)
        linv[r] = 1.f / (__shfl(lrun, r + hi8, 32) + 1e-13f);
    float* xb = X + ((size_t)b * N_ + (itile << 4)) * D_ + h * H_;
#pragma unroll
    for (int r = 0; r < 8; ++r) {
        size_t ro = (size_t)(r + hi8) * D_;
        xb[ro + 0  + lrow16] = acc0[r] * linv[r];
        xb[ro + 16 + lrow16] = acc1[r] * linv[r];
        xb[ro + 32 + lrow16] = acc2[r] * linv[r];
        xb[ro + 48 + lrow16] = acc3[r] * linv[r];
    }
}

// ---------------------------------------------------------------- host orchestration
extern "C" void kernel_launch(void* const* d_in, const int* in_sizes, int n_in,
                              void* d_out, int out_size, void* d_ws, size_t ws_size,
                              hipStream_t stream) {
    const float* node_emb = (const float*)d_in[0];
    const float* q_emb    = (const float*)d_in[1];
    const float* W_fc     = (const float*)d_in[2];
    const float* w_dc     = (const float*)d_in[3];
    const float* dense_w  = (const float*)d_in[4];
    const float* ln_in_g  = (const float*)d_in[5];
    const float* ln_in_b  = (const float*)d_in[6];
    const float* W_qv     = (const float*)d_in[7];
    const float* W_kv     = (const float*)d_in[8];
    const float* W_vv     = (const float*)d_in[9];
    const float* W_qc     = (const float*)d_in[10];
    const float* W_kc     = (const float*)d_in[11];
    const float* W_vc     = (const float*)d_in[12];
    const float* ln_q_g   = (const float*)d_in[13];
    const float* ln_q_b   = (const float*)d_in[14];
    const float* ln_k_g   = (const float*)d_in[15];
    const float* ln_k_b   = (const float*)d_in[16];
    const float* ln_v_g   = (const float*)d_in[17];
    const float* ln_v_b   = (const float*)d_in[18];
    const float* W_att    = (const float*)d_in[19];
    const float* W_u      = (const float*)d_in[20];
    const float* ln_out_g = (const float*)d_in[21];
    const float* ln_out_b = (const float*)d_in[22];
    const int* mask_e     = (const int*)d_in[23];
    const int* mask_t     = (const int*)d_in[24];

    char* ws = (char*)d_ws;
    size_t cur = 0;
    auto alloc = [&](size_t bytes) -> char* {
        char* p = ws + cur;
        cur += (bytes + 255) & ~(size_t)255;
        return p;
    };
    bf16_t* dense_p = (bf16_t*)alloc((size_t)D2_ * D2_ * 2);
    bf16_t* qv_p    = (bf16_t*)alloc((size_t)D2_ * D_ * 2);
    bf16_t* kv_p    = (bf16_t*)alloc((size_t)D2_ * D_ * 2);
    bf16_t* vv_p    = (bf16_t*)alloc((size_t)D2_ * D_ * 2);
    bf16_t* u_p     = (bf16_t*)alloc((size_t)D2_ * D_ * 2);
    float* h1   = (float*)alloc((size_t)B_ * D2_ * 4);
    float* gq   = (float*)alloc((size_t)T_ * B_ * D_ * 4);
    float* gk   = (float*)alloc((size_t)T_ * B_ * D_ * 4);
    float* gv   = (float*)alloc((size_t)T_ * B_ * D_ * 4);
    signed char* codes = (signed char*)alloc((size_t)B_ * N_ * N_);
    bf16_t* catb  = (bf16_t*)alloc((size_t)M_ * D2_ * 2);
    float* ie_raw = (float*)alloc((size_t)M_ * D2_ * 4);
    bf16_t* ie_b  = (bf16_t*)alloc((size_t)M_ * D2_ * 2);
    float* q_raw  = (float*)alloc((size_t)M_ * D_ * 4);
    float* k_raw  = (float*)alloc((size_t)M_ * D_ * 4);
    float* v_raw  = (float*)alloc((size_t)M_ * D_ * 4);
    float* xq     = (float*)alloc((size_t)M_ * D_ * 4);
    float* xk     = (float*)alloc((size_t)M_ * D_ * 4);
    float* xv     = (float*)alloc((size_t)M_ * D_ * 4);
    bf16_t* xvt   = (bf16_t*)alloc((size_t)B_ * NH_ * H_ * N_ * 2);
    float* asrc   = (float*)alloc((size_t)9 * B_ * NH_ * N_ * 4);
    float* adst   = (float*)alloc((size_t)9 * B_ * NH_ * N_ * 4);
    float* Xbuf   = (float*)alloc((size_t)M_ * D_ * 4);
    float* u_raw  = (float*)alloc((size_t)M_ * D_ * 4);
    float* enc    = (float*)alloc((size_t)M_ * D_ * 4);
    (void)ws_size; (void)in_sizes; (void)n_in;

    const int TB = 256;
    // ---- one-time prep
    k_pack_b<<<(32 * 64 * 32 + TB - 1) / TB, TB, 0, stream>>>(dense_w, dense_p, D2_, D2_);
    k_pack_b<<<(32 * 32 * 32 + TB - 1) / TB, TB, 0, stream>>>(W_qv, qv_p, D2_, D_);
    k_pack_b<<<(32 * 32 * 32 + TB - 1) / TB, TB, 0, stream>>>(W_kv, kv_p, D2_, D_);
    k_pack_b<<<(32 * 32 * 32 + TB - 1) / TB, TB, 0, stream>>>(W_vv, vv_p, D2_, D_);
    k_pack_b<<<(32 * 32 * 32 + TB - 1) / TB, TB, 0, stream>>>(W_u, u_p, D2_, D_);
    k_h1<<<B_, TB, 0, stream>>>(q_emb, W_fc, h1);
    k_gates<<<T_ * B_, TB, 0, stream>>>(h1, w_dc, W_qc, W_kc, W_vc, gq, gk, gv);
    {
        size_t total = (size_t)B_ * N_ * N_;
        k_codes<<<(int)((total + TB - 1) / TB), TB, 0, stream>>>(mask_e, mask_t, codes);
    }
    k_copy_f32<<<(M_ * D_ + TB - 1) / TB, TB, 0, stream>>>(node_emb, enc, M_ * D_);

    const int catBlocks = (int)(((size_t)M_ * D2_ + TB - 1) / TB);
    for (int t = 0; t < T_; ++t) {
        // ie = LN(cat(enc, node_emb) @ dense_w)
        k_cat_bf16<<<catBlocks, TB, 0, stream>>>(enc, node_emb, catb);
        k_gemm_bf16<<<(256 * 16) / 8, TB, 0, stream>>>(catb, dense_p, ie_raw, 256, 32, D2_, nullptr, N_);
        k_ln<<<M_, TB, 0, stream>>>(ie_raw, ln_in_g, ln_in_b, nullptr, ie_b, D2_);
        // gated q/k/v projections + LN
        k_gemm_bf16<<<(256 * 8) / 8, TB, 0, stream>>>(ie_b, qv_p, q_raw, 256, 32, D_, gq + (size_t)t * B_ * D_, N_);
        k_gemm_bf16<<<(256 * 8) / 8, TB, 0, stream>>>(ie_b, kv_p, k_raw, 256, 32, D_, gk + (size_t)t * B_ * D_, N_);
        k_gemm_bf16<<<(256 * 8) / 8, TB, 0, stream>>>(ie_b, vv_p, v_raw, 256, 32, D_, gv + (size_t)t * B_ * D_, N_);
        k_ln<<<M_, TB, 0, stream>>>(q_raw, ln_q_g, ln_q_b, xq, nullptr, D_);
        k_ln<<<M_, TB, 0, stream>>>(k_raw, ln_k_g, ln_k_b, xk, nullptr, D_);
        k_ln<<<M_, TB, 0, stream>>>(v_raw, ln_v_g, ln_v_b, xv, nullptr, D_);
        // additive attention rank-1 terms + V transpose
        k_att_proj<<<(2 * 9 * B_ * NH_ * N_) / TB, TB, 0, stream>>>(xq, xk, W_att, asrc, adst);
        {
            size_t total = (size_t)B_ * NH_ * H_ * N_;
            k_xvt<<<(int)((total + TB - 1) / TB), TB, 0, stream>>>(xv, xvt);
        }
        // flash attention: 2048 waves
        k_attn<<<(B_ * NH_ * 64) / 8, TB, 0, stream>>>(codes, asrc, adst, xvt, Xbuf);
        // node_enc = LN(cat(enc, X) @ W_u)
        k_cat_bf16<<<catBlocks, TB, 0, stream>>>(enc, Xbuf, catb);
        k_gemm_bf16<<<(256 * 8) / 8, TB, 0, stream>>>(catb, u_p, u_raw, 256, 32, D_, nullptr, N_);
        k_ln<<<M_, TB, 0, stream>>>(u_raw, ln_out_g, ln_out_b, enc, nullptr, D_);
    }
    k_copy_f32<<<(M_ * D_ + TB - 1) / TB, TB, 0, stream>>>(enc, (float*)d_out, M_ * D_);
}
